// GraphAttentionCPU_30159260352621
// MI455X (gfx1250) — compile-verified, matrix operand order checked by reading, not run on hardware
//
#include <hip/hip_runtime.h>
#include <hip/hip_bf16.h>

#define D 128
#define NEG_SLOPE 0.2f
#define BN_EPS 1e-5f

typedef __attribute__((ext_vector_type(16))) _Float16 v16h;
typedef __attribute__((ext_vector_type(8)))  float    v8f;

// ---------------- init: zero y accumulator (d_out), init segment-max bits to -inf, denom to 0
__global__ void gat_init(float* __restrict__ y, unsigned* __restrict__ maxbits,
                         float* __restrict__ denom, int n_nodes, long total) {
  long i = (long)blockIdx.x * blockDim.x + threadIdx.x;
  const long stride = (long)gridDim.x * blockDim.x;
  for (; i < total; i += stride) {
    y[i] = 0.0f;
    if (i < n_nodes) { maxbits[i] = 0xFF800000u; denom[i] = 0.0f; }
  }
}

// ---------------- pack W (f32 row-major 128x128) into per-lane f16 B-fragment layout.
// Fragment packing per CDNA5 ISA 7.12.2 (wave32, 16-bit B 32x16):
//   lane L: half = L>>4, r = L&15; col = nt*16 + r
//   elems j<8 -> k = kb*32 + half*8 + j ; j>=8 -> k = kb*32 + half*8 + 16 + (j-8)
// Wpack[((nt*4 + kb)*32 + lane)*16 + j]  (8 nt-tiles * 4 kb * 32 lanes * 16 f16 = 32 KB)
__global__ void gat_pack_w(const float* __restrict__ W, _Float16* __restrict__ Wpack) {
  const int idx = blockIdx.x * blockDim.x + threadIdx.x;   // [0, 1024)
  if (idx >= 8 * 4 * 32) return;
  const int lane = idx & 31;
  const int kb   = (idx >> 5) & 3;
  const int nt   = idx >> 7;
  const int half = lane >> 4;
  const int r    = lane & 15;
  const int ncol = nt * 16 + r;
  const int k0   = kb * 32 + half * 8;
  _Float16* out = Wpack + (size_t)idx * 16;
#pragma unroll
  for (int j = 0; j < 8; ++j) out[j]     = (_Float16)W[(k0 + j) * D + ncol];
#pragma unroll
  for (int j = 0; j < 8; ++j) out[j + 8] = (_Float16)W[(k0 + 16 + j) * D + ncol];
}

// ---------------- h = x @ W via v_wmma_f32_16x16x32_f16
// One block = 256 threads = 8 waves; block handles 16 rows, wave w handles cols [16w,16w+16).
// B fragments come pre-packed (one 32B vector load each); A loaded f32 + cvt.
__global__ __launch_bounds__(256) void gat_gemm_wmma(const float* __restrict__ x,
                                                     const _Float16* __restrict__ Wpack,
                                                     float* __restrict__ h,
                                                     int n_nodes) {
  const int lane = threadIdx.x & 31;
  const int wave = threadIdx.x >> 5;   // N-tile 0..7
  const int half = lane >> 4;
  const int r    = lane & 15;
  const int rowBlock = blockIdx.x * 16;

  int arow = rowBlock + r;
  if (arow >= n_nodes) arow = n_nodes - 1;   // clamp (no divergence; EXEC stays full)
  const int ncol = wave * 16 + r;

  const v16h* wp = (const v16h*)Wpack + (size_t)wave * 4 * 32 + lane;

  v8f acc = {};
#pragma unroll
  for (int kb = 0; kb < 4; ++kb) {
    const int k0 = kb * 32 + half * 8;
    const float* ap = x + (long)arow * D + k0;

    v16h a;
    {
      const float4 lo0 = *(const float4*)(ap + 0);
      const float4 lo1 = *(const float4*)(ap + 4);
      const float4 hi0 = *(const float4*)(ap + 16);
      const float4 hi1 = *(const float4*)(ap + 20);
      a[0]  = (_Float16)lo0.x; a[1]  = (_Float16)lo0.y; a[2]  = (_Float16)lo0.z; a[3]  = (_Float16)lo0.w;
      a[4]  = (_Float16)lo1.x; a[5]  = (_Float16)lo1.y; a[6]  = (_Float16)lo1.z; a[7]  = (_Float16)lo1.w;
      a[8]  = (_Float16)hi0.x; a[9]  = (_Float16)hi0.y; a[10] = (_Float16)hi0.z; a[11] = (_Float16)hi0.w;
      a[12] = (_Float16)hi1.x; a[13] = (_Float16)hi1.y; a[14] = (_Float16)hi1.z; a[15] = (_Float16)hi1.w;
    }
    const v16h b = wp[kb * 32];   // 32-byte vector load: 2x global_load_b128

    acc = __builtin_amdgcn_wmma_f32_16x16x32_f16(false, a, false, b,
                                                 (short)0, acc, false, false);
  }

#pragma unroll
  for (int v = 0; v < 8; ++v) {
    const int row = rowBlock + half * 8 + v;
    if (row < n_nodes) h[(long)row * D + ncol] = acc[v];
  }
}

// ---------------- per-node attention halves: a_s = h.att_src, a_d = h.att_dst (wave per node)
__global__ __launch_bounds__(256) void gat_node_attn(const float* __restrict__ h,
                                                     const float* __restrict__ att_src,
                                                     const float* __restrict__ att_dst,
                                                     float* __restrict__ a_s,
                                                     float* __restrict__ a_d, int n_nodes) {
  const int node = (int)(((long)blockIdx.x * blockDim.x + threadIdx.x) >> 5);
  const int lane = threadIdx.x & 31;
  if (node >= n_nodes) return;
  const float* hp = h + (long)node * D;
  float s = 0.f, d = 0.f;
#pragma unroll
  for (int c = 0; c < D; c += 32) {
    const float v = hp[c + lane];
    s += v * att_src[c + lane];
    d += v * att_dst[c + lane];
  }
#pragma unroll
  for (int off = 16; off > 0; off >>= 1) {
    s += __shfl_down(s, off, 32);
    d += __shfl_down(d, off, 32);
  }
  if (lane == 0) { a_s[node] = s; a_d[node] = d; }
}

// ---------------- edge logits + segment max (sign-split int/uint atomic trick)
__global__ void gat_edge_logit_max(const int* __restrict__ ei,
                                   const float* __restrict__ a_s,
                                   const float* __restrict__ a_d,
                                   float* __restrict__ elog,
                                   unsigned* __restrict__ maxbits, int n_edges) {
  const int e = blockIdx.x * blockDim.x + threadIdx.x;
  if (e >= n_edges) return;
  const int src = ei[e];
  const int dst = ei[n_edges + e];
  float v = a_s[src] + a_d[dst];
  v = (v > 0.f) ? v : NEG_SLOPE * v;
  elog[e] = v;
  if (v >= 0.f) atomicMax((int*)(maxbits + dst), __float_as_int(v));
  else          atomicMin(maxbits + dst, __float_as_uint(v));
}

// ---------------- p = exp(e - max[dst]); denom[dst] += p
__global__ void gat_edge_exp_sum(const int* __restrict__ ei,
                                 const float* __restrict__ elog,
                                 const unsigned* __restrict__ maxbits,
                                 float* __restrict__ p,
                                 float* __restrict__ denom, int n_edges) {
  const int e = blockIdx.x * blockDim.x + threadIdx.x;
  if (e >= n_edges) return;
  const int dst = ei[n_edges + e];
  const float m = __uint_as_float(maxbits[dst]);   // finite for any dst with edges
  const float pv = __expf(elog[e] - m);
  p[e] = pv;
  atomicAdd(denom + dst, pv);
}

// ---------------- y[dst] += alpha * h[src]  (wave per edge; h is L2-resident)
__global__ __launch_bounds__(256) void gat_aggregate(const int* __restrict__ ei,
                                                     const float* __restrict__ p,
                                                     const float* __restrict__ denom,
                                                     const float* __restrict__ h,
                                                     float* __restrict__ y, int n_edges) {
  const int e = (int)(((long)blockIdx.x * blockDim.x + threadIdx.x) >> 5);
  const int lane = threadIdx.x & 31;
  if (e >= n_edges) return;
  const int src = ei[e];
  const int dst = ei[n_edges + e];
  const float alpha = p[e] / denom[dst];
  const float4 v = ((const float4*)(h + (long)src * D))[lane];
  float* yp = y + (long)dst * D + lane * 4;
  atomicAdd(yp + 0, alpha * v.x);
  atomicAdd(yp + 1, alpha * v.y);
  atomicAdd(yp + 2, alpha * v.z);
  atomicAdd(yp + 3, alpha * v.w);
}

// ---------------- out = x + relu(gamma * (y+bias)/sqrt(1+eps) + beta)   (in-place on y==d_out)
__global__ void gat_finalize(const float* __restrict__ x,
                             const float* __restrict__ bias,
                             const float* __restrict__ gamma,
                             const float* __restrict__ beta,
                             float* __restrict__ y, long total) {
  const float inv = rsqrtf(1.0f + BN_EPS);
  long i = (long)blockIdx.x * blockDim.x + threadIdx.x;
  const long stride = (long)gridDim.x * blockDim.x;
  for (; i < total; i += stride) {
    const int d = (int)(i & (D - 1));
    float v = (y[i] + bias[d]) * inv;
    v = gamma[d] * v + beta[d];
    v = (v > 0.f) ? v : 0.f;
    y[i] = x[i] + v;
  }
}

extern "C" void kernel_launch(void* const* d_in, const int* in_sizes, int n_in,
                              void* d_out, int out_size, void* d_ws, size_t ws_size,
                              hipStream_t stream) {
  const float* x       = (const float*)d_in[0];
  const int*   ei      = (const int*)  d_in[1];
  const float* W       = (const float*)d_in[2];
  const float* att_src = (const float*)d_in[3];
  const float* att_dst = (const float*)d_in[4];
  const float* bias    = (const float*)d_in[5];
  const float* gamma   = (const float*)d_in[6];
  const float* beta    = (const float*)d_in[7];

  const int n_nodes = in_sizes[0] / D;
  const int n_edges = in_sizes[1] / 2;

  float* y = (float*)d_out;                    // y accumulator lives in d_out

  // workspace layout (~33 MB)
  float*    h       = (float*)d_ws;                          // N*D
  float*    a_s     = h + (size_t)n_nodes * D;               // N
  float*    a_d     = a_s + n_nodes;                         // N
  float*    elog    = a_d + n_nodes;                         // E
  float*    p       = elog + n_edges;                        // E
  unsigned* maxbits = (unsigned*)(p + n_edges);              // N
  float*    denom   = (float*)(maxbits + n_nodes);           // N
  _Float16* Wpack   = (_Float16*)(denom + n_nodes);          // 8*4*32*16 f16 = 32 KB

  const long total = (long)n_nodes * D;

  gat_init<<<2048, 256, 0, stream>>>(y, maxbits, denom, n_nodes, total);
  gat_pack_w<<<4, 256, 0, stream>>>(W, Wpack);
  gat_gemm_wmma<<<(n_nodes + 15) / 16, 256, 0, stream>>>(x, Wpack, h, n_nodes);
  gat_node_attn<<<(n_nodes + 7) / 8, 256, 0, stream>>>(h, att_src, att_dst, a_s, a_d, n_nodes);
  gat_edge_logit_max<<<(n_edges + 255) / 256, 256, 0, stream>>>(ei, a_s, a_d, elog, maxbits, n_edges);
  gat_edge_exp_sum<<<(n_edges + 255) / 256, 256, 0, stream>>>(ei, elog, maxbits, p, denom, n_edges);
  gat_aggregate<<<(n_edges + 7) / 8, 256, 0, stream>>>(ei, p, denom, h, y, n_edges);
  gat_finalize<<<4096, 256, 0, stream>>>(x, bias, gamma, beta, y, total);
}
